// MoE_layer_flux_26044681683727
// MI455X (gfx1250) — matrix operand order repaired and checked by reading, compiled, not run on hardware
//
#include <hip/hip_runtime.h>
#include <hip/hip_bf16.h>

// ---------------- Problem constants (match reference) ----------------
#define NTOKENS 8192
#define TOPK    2
#define HID     1024
#define FFN     4096
#define NEXP    8
#define MTOT    (NTOKENS * TOPK)   // 16384 dispatched rows
#define CAP     (MTOT / NEXP)      // 2048 rows per expert

// ---------------- Types ----------------
typedef __bf16 bf16_t;
typedef __attribute__((ext_vector_type(16))) __bf16 bf16x16;
typedef __attribute__((ext_vector_type(8)))  __bf16 bf16x8;
typedef __attribute__((ext_vector_type(4)))  __bf16 bf16x4;
typedef __attribute__((ext_vector_type(8)))  float  f32x8;
typedef __attribute__((ext_vector_type(4)))  unsigned int u32x4;
typedef __attribute__((ext_vector_type(8)))  int i32x8;
typedef __attribute__((ext_vector_type(4)))  int i32x4;

union BFrag { bf16x16 v; bf16x8 h[2]; };

__device__ __forceinline__ bf16_t f2bf(float f) {
    union { float f; unsigned int u; } in; in.f = f;
    unsigned int u = in.u;
    u += 0x7FFFu + ((u >> 16) & 1u);           // round-to-nearest-even
    union { unsigned short s; bf16_t b; } out;
    out.s = (unsigned short)(u >> 16);
    return out.b;
}

// Branch-free exact-GELU: 0.5*x*(1+erf(x/sqrt(2))) with Abramowitz-Stegun 7.1.26
// erf approximation (|err| < 1.5e-7, far below bf16 quantization noise).
__device__ __forceinline__ float gelu_erf(float x) {
    const float z  = x * 0.70710678118654752f;
    const float az = __builtin_fabsf(z);
#if __has_builtin(__builtin_amdgcn_rcpf)
    const float t = __builtin_amdgcn_rcpf(1.0f + 0.3275911f * az);
#else
    const float t = 1.0f / (1.0f + 0.3275911f * az);
#endif
    float poly = t * (0.254829592f +
                 t * (-0.284496736f +
                 t * (1.421413741f +
                 t * (-1.453152027f +
                 t *   1.061405429f))));
    float er = 1.0f - poly * __expf(-az * az);
    er = __builtin_copysignf(er, z);
    return 0.5f * x * (1.0f + er);
}

// ---------------- Kernel 1: AG-scatter, fp32 -> bf16 ----------------
__global__ __launch_bounds__(256)
void scatter_kernel(const float* __restrict__ x, const int* __restrict__ sidx,
                    bf16_t* __restrict__ disp) {
    const int tk   = blockIdx.x;           // t*TOPK + k
    const int t    = tk >> 1;
    const int row  = sidx[tk];
    const int c    = threadIdx.x * 4;
    const float4 f = *reinterpret_cast<const float4*>(&x[(size_t)t * HID + c]);
    bf16x4 v;
    v[0] = f2bf(f.x); v[1] = f2bf(f.y); v[2] = f2bf(f.z); v[3] = f2bf(f.w);
    *reinterpret_cast<bf16x4*>(&disp[(size_t)row * HID + c]) = v;
}

// ---------------- Kernel 2: weight convert + transpose ----------------
// in: f32 [E][R][C]  ->  out: bf16 [E][C][R]   (32x32 tiles via LDS)
template<int R, int C>
__global__ __launch_bounds__(256)
void transpose_bf16_kernel(const float* __restrict__ in, bf16_t* __restrict__ out) {
    const int e = blockIdx.z;
    in  += (size_t)e * R * C;
    out += (size_t)e * R * C;
    const unsigned r0 = blockIdx.y * 32;
    const unsigned c0 = blockIdx.x * 32;
    __shared__ bf16_t t[32][33];

    const unsigned tid = threadIdx.x;
    const unsigned rr  = tid >> 3;        // 0..31
    const unsigned cq  = tid & 7;         // 0..7 (4-col segment)

    const float4 f = *reinterpret_cast<const float4*>(&in[(size_t)(r0 + rr) * C + c0 + cq * 4]);
    t[rr][cq * 4 + 0] = f2bf(f.x);
    t[rr][cq * 4 + 1] = f2bf(f.y);
    t[rr][cq * 4 + 2] = f2bf(f.z);
    t[rr][cq * 4 + 3] = f2bf(f.w);
    __syncthreads();

    bf16x4 v;
    v[0] = t[cq * 4 + 0][rr];
    v[1] = t[cq * 4 + 1][rr];
    v[2] = t[cq * 4 + 2][rr];
    v[3] = t[cq * 4 + 3][rr];
    *reinterpret_cast<bf16x4*>(&out[(size_t)(c0 + rr) * R + r0 + cq * 4]) = v;
}

// ---------------- Kernel 3/4: grouped WMMA GEMM (per-expert) ----------------
// C[e] = act(A[e] (MxK bf16, row-major) * Bt[e] (NxK bf16, row-major)^T)
// Workgroup tile 256x128, BK=32, double-buffered LDS.
// 8 waves in 4x2: each wave owns a 64x64 tile = 4x4 WMMA frags (16 wmma / K-step).
// Tiles are staged by the Tensor Data Mover (wave 0 issues D# descriptors;
// completion enforced with s_wait_tensorcnt before the workgroup barrier).
#define LDSTR 40   // LDS row stride in bf16 (64B data + 16B TDM pad; 16B-aligned)

#define HAVE_TDM __has_builtin(__builtin_amdgcn_tensor_load_to_lds)

#if HAVE_TDM
// Issue one 2D TDM load: tile (tileRows x 32 bf16) from row-major source
// (row stride kElems) at gsrc into LDS at ldsOff, padded to LDSTR*2 B rows.
// D# layout per CDNA5 ISA §8.3 (group0) / §8.4 (group1).
__device__ __forceinline__ void tdm_load_tile(const bf16_t* gsrc, unsigned ldsOff,
                                              unsigned kElems, unsigned tensorRows,
                                              unsigned tileRows) {
    const unsigned long long ga = (unsigned long long)(const void*)gsrc;
    u32x4 g0;
    g0[0] = 1u;                                          // count=1 (valid user D#)
    g0[1] = ldsOff;                                      // lds_addr (bytes)
    g0[2] = (unsigned)ga;                                // global_addr[31:0]
    g0[3] = (unsigned)((ga >> 32) & 0x01FFFFFFull)       // global_addr[56:32]
          | (2u << 30);                                  // type=2 ("image")
    i32x8 g1;
    g1[0] = (int)((1u << 16)                             // data_size = 2 bytes
                | (1u << 20)                             // pad_enable
                | (3u << 22)                             // pad_interval = 16 DWORDs (64B)
                | (3u << 25));                           // pad_amount   =  4 DWORDs (16B)
    g1[1] = (int)((kElems & 0xFFFFu) << 16);             // tensor_dim0[15:0]
    g1[2] = (int)((kElems >> 16)                         // tensor_dim0[31:16]
                | ((tensorRows & 0xFFFFu) << 16));       // tensor_dim1[15:0]
    g1[3] = (int)((tensorRows >> 16)                     // tensor_dim1[31:16]
                | (32u << 16));                          // tile_dim0 = 32 elements
    g1[4] = (int)tileRows;                               // tile_dim1 (tile_dim2 = 0)
    g1[5] = (int)kElems;                                 // tensor_dim0_stride[31:0]
    g1[6] = 0;                                           // stride hi / dim1_stride
    g1[7] = 0;
    const i32x4 gz4 = {0, 0, 0, 0};                      // groups 2/3 unused (<=2D)
    const i32x8 gz8 = {0, 0, 0, 0, 0, 0, 0, 0};
    __builtin_amdgcn_tensor_load_to_lds(g0, g1, gz4, gz4, gz8, 0);
}
#endif

// Fallback manual staging (also used if TDM builtin is unavailable)
template<int K, int ROWS>
__device__ __forceinline__ void stage_tile(const bf16_t* __restrict__ src, unsigned rowBase,
                                           unsigned k0, bf16_t* __restrict__ lds, unsigned tid) {
#pragma unroll
    for (int it = 0; it < ROWS / 64; ++it) {
        const unsigned c = tid + it * 256;
        const unsigned r = c >> 2, q = c & 3;
        *reinterpret_cast<bf16x8*>(&lds[r * LDSTR + q * 8]) =
            *reinterpret_cast<const bf16x8*>(&src[(size_t)(rowBase + r) * K + k0 + q * 8]);
    }
}

template<int M, int N, int K, bool GELU_BF16_OUT>
__global__ __launch_bounds__(256)
void wmma_gemm_kernel(const bf16_t* __restrict__ Aall, const bf16_t* __restrict__ Btall,
                      void* __restrict__ Cout) {
    constexpr int BM = 256, BN = 128;
    const int e = blockIdx.z;
    const bf16_t* A  = Aall  + (size_t)e * M * K;
    const bf16_t* Bt = Btall + (size_t)e * N * K;
    const unsigned rowBase = blockIdx.y * BM;
    const unsigned colBase = blockIdx.x * BN;

    __shared__ __align__(16) bf16_t ldsA[2][BM * LDSTR];   // 2 x 20480 B
    __shared__ __align__(16) bf16_t ldsB[2][BN * LDSTR];   // 2 x 10240 B  -> 60 KB total

    const unsigned tid   = threadIdx.x;
    const unsigned lane  = tid & 31;
    const unsigned wave  = tid >> 5;        // 0..7
    const unsigned waveM = wave >> 1;       // 0..3  (64 rows each)
    const unsigned waveN = wave & 1;        // 0..1  (64 cols each)

    f32x8 acc[4][4];
    {
        f32x8 z = {};
        for (int i = 0; i < 4; ++i)
            for (int j = 0; j < 4; ++j)
                acc[i][j] = z;
    }

    // Fragment LDS offsets (constant across K loop)
    const unsigned mOff = (waveM * 64 + (lane & 15)) * LDSTR + (lane >> 4) * 8;
    const unsigned nOff = (waveN * 64 + (lane & 15)) * LDSTR + (lane >> 4) * 8;

    // ---- prologue: stage tile 0 ----
#if HAVE_TDM
    if (wave == 0) {
        tdm_load_tile(&A [(size_t)rowBase * K], (unsigned)(size_t)&ldsA[0][0], K, M - rowBase, BM);
        tdm_load_tile(&Bt[(size_t)colBase * K], (unsigned)(size_t)&ldsB[0][0], K, N - colBase, BN);
        __builtin_amdgcn_s_wait_tensorcnt(0);
    }
#else
    stage_tile<K, BM>(A,  rowBase, 0, ldsA[0], tid);
    stage_tile<K, BN>(Bt, colBase, 0, ldsB[0], tid);
#endif
    __syncthreads();

    unsigned cur = 0;
#pragma unroll 1
    for (unsigned k0 = 0; k0 < K; k0 += 32) {
        // stage next tile into the other buffer (overlaps with WMMA below)
        if (k0 + 32 < K) {
#if HAVE_TDM
            if (wave == 0) {
                tdm_load_tile(&A [(size_t)rowBase * K + k0 + 32],
                              (unsigned)(size_t)&ldsA[cur ^ 1][0], K, M - rowBase, BM);
                tdm_load_tile(&Bt[(size_t)colBase * K + k0 + 32],
                              (unsigned)(size_t)&ldsB[cur ^ 1][0], K, N - colBase, BN);
            }
#else
            stage_tile<K, BM>(A,  rowBase, k0 + 32, ldsA[cur ^ 1], tid);
            stage_tile<K, BN>(Bt, colBase, k0 + 32, ldsB[cur ^ 1], tid);
#endif
        }

        // fragments (ISA 7.12.2 16-bit layouts)
        BFrag afrag[4], bfrag[4];
#pragma unroll
        for (int i = 0; i < 4; ++i) {
            const unsigned off = mOff + i * 16 * LDSTR;
            afrag[i].h[0] = *reinterpret_cast<const bf16x8*>(&ldsA[cur][off]);
            afrag[i].h[1] = *reinterpret_cast<const bf16x8*>(&ldsA[cur][off + 16]);
        }
#pragma unroll
        for (int j = 0; j < 4; ++j) {
            const unsigned off = nOff + j * 16 * LDSTR;
            bfrag[j].h[0] = *reinterpret_cast<const bf16x8*>(&ldsB[cur][off]);
            bfrag[j].h[1] = *reinterpret_cast<const bf16x8*>(&ldsB[cur][off + 16]);
        }

#pragma unroll
        for (int i = 0; i < 4; ++i)
#pragma unroll
            for (int j = 0; j < 4; ++j)
                acc[i][j] = __builtin_amdgcn_wmma_f32_16x16x32_bf16(
                    false, afrag[i].v, false, bfrag[j].v,
                    (short)0, acc[i][j], false, false);

#if HAVE_TDM
        if (wave == 0) __builtin_amdgcn_s_wait_tensorcnt(0);
#endif
        __syncthreads();
        cur ^= 1;
    }

    // ---- epilogue: C/D layout -> global ----
    const unsigned lhalf = lane >> 4;       // 0: rows v, 1: rows v+8
    const unsigned lcol  = lane & 15;
#pragma unroll
    for (int i = 0; i < 4; ++i) {
#pragma unroll
        for (int j = 0; j < 4; ++j) {
            const unsigned col = colBase + waveN * 64 + j * 16 + lcol;
#pragma unroll
            for (int v = 0; v < 8; ++v) {
                const unsigned row = rowBase + waveM * 64 + i * 16 + lhalf * 8 + v;
                float val = acc[i][j][v];
                if constexpr (GELU_BF16_OUT) {
                    val = gelu_erf(val);
                    bf16_t* C = (bf16_t*)Cout + (size_t)e * M * N;
                    C[(size_t)row * N + col] = f2bf(val);
                } else {
                    float* C = (float*)Cout + (size_t)e * M * N;
                    C[(size_t)row * N + col] = val;
                }
            }
        }
    }
}

// ---------------- Kernel 5: gather-RS combine ----------------
__global__ __launch_bounds__(256)
void gather_kernel(const float* __restrict__ ye, const int* __restrict__ sidx,
                   float* __restrict__ out) {
    const int t  = blockIdx.x;
    const int c  = threadIdx.x * 4;
    const int r0 = sidx[t * 2 + 0];
    const int r1 = sidx[t * 2 + 1];
    const float4 a = *reinterpret_cast<const float4*>(&ye[(size_t)r0 * HID + c]);
    const float4 b = *reinterpret_cast<const float4*>(&ye[(size_t)r1 * HID + c]);
    float4 o;
    o.x = a.x + b.x; o.y = a.y + b.y; o.z = a.z + b.z; o.w = a.w + b.w;
    *reinterpret_cast<float4*>(&out[(size_t)t * HID + c]) = o;
}

// ---------------- Host launcher ----------------
extern "C" void kernel_launch(void* const* d_in, const int* in_sizes, int n_in,
                              void* d_out, int out_size, void* d_ws, size_t ws_size,
                              hipStream_t stream) {
    const float* x    = (const float*)d_in[0];   // [NTOKENS, HID]
    const float* w0   = (const float*)d_in[1];   // [E, HID, FFN]
    const float* w1   = (const float*)d_in[2];   // [E, FFN, HID]
    const int*   sidx = (const int*)  d_in[3];   // [NTOKENS, TOPK]
    // d_in[4] = splits (balanced by construction; unused)
    float* out = (float*)d_out;

    // Workspace layout (bytes):
    //   disp bf16 [MTOT][HID]   :  32 MB
    //   mid  bf16 [MTOT][FFN]   : 128 MB
    //   ye   f32  [MTOT][HID]   :  64 MB
    //   w0t  bf16 [E][FFN][HID] :  64 MB
    //   w1t  bf16 [E][HID][FFN] :  64 MB
    char* ws = (char*)d_ws;
    size_t off = 0;
    bf16_t* disp = (bf16_t*)(ws + off); off += (size_t)MTOT * HID * 2;
    bf16_t* mid  = (bf16_t*)(ws + off); off += (size_t)MTOT * FFN * 2;
    float*  ye   = (float*)(ws + off);  off += (size_t)MTOT * HID * 4;
    bf16_t* w0t  = (bf16_t*)(ws + off); off += (size_t)NEXP * HID * FFN * 2;
    bf16_t* w1t  = (bf16_t*)(ws + off);

    // 1) weight convert + transpose (fp32 -> bf16, [K][N] -> [N][K])
    transpose_bf16_kernel<HID, FFN><<<dim3(FFN / 32, HID / 32, NEXP), 256, 0, stream>>>(w0, w0t);
    transpose_bf16_kernel<FFN, HID><<<dim3(HID / 32, FFN / 32, NEXP), 256, 0, stream>>>(w1, w1t);

    // 2) AG-scatter with fp32->bf16 demotion
    scatter_kernel<<<MTOT, 256, 0, stream>>>(x, sidx, disp);

    // 3) grouped GEMM0 + exact-erf GELU -> mid (bf16)
    wmma_gemm_kernel<CAP, FFN, HID, true>
        <<<dim3(FFN / 128, CAP / 256, NEXP), 256, 0, stream>>>(disp, w0t, (void*)mid);

    // 4) grouped GEMM1 -> ye (f32)
    wmma_gemm_kernel<CAP, HID, FFN, false>
        <<<dim3(HID / 128, CAP / 256, NEXP), 256, 0, stream>>>(mid, w1t, (void*)ye);

    // 5) gather-RS combine
    gather_kernel<<<NTOKENS, 256, 0, stream>>>(ye, sidx, out);
}
